// ACRGNN_19688130085206
// MI455X (gfx1250) — matrix operand order repaired and checked by reading, not compiled, not run on hardware
//
#include <hip/hip_runtime.h>
#include <hip/hip_bf16.h>

#define NN   50000
#define EE   800000
#define HH   128
#define OUTD 64
#define LL   3
#define GG   64
#define EPSV 1e-5f

typedef __attribute__((ext_vector_type(16))) __bf16 v16bf;
typedef __attribute__((ext_vector_type(8)))  float  v8f;

union Frag16 {
    unsigned int u[8];
    v16bf b;
};

// ---------- scalar bf16 helpers ----------
__device__ __forceinline__ unsigned short f2bf(float f) {
    unsigned int u = __float_as_uint(f);
    u += 0x7fffu + ((u >> 16) & 1u);           // round-to-nearest-even
    return (unsigned short)(u >> 16);
}
__device__ __forceinline__ float bflo2f(unsigned int p) { return __uint_as_float(p << 16); }
__device__ __forceinline__ float bfhi2f(unsigned int p) { return __uint_as_float(p & 0xffff0000u); }

// ---------- WMMA fragment loaders (bf16, 16x16x32) ----------
// A 16x32 (MxK), row-major source, rowp = dword ptr to this lane's row (M = lane&15).
// ka = kTile + ((lane>>4)<<4).  VGPR0..3 hold K=ka..ka+7, VGPR4..7 hold K=ka+8..ka+15.
__device__ __forceinline__ v16bf load_a(const unsigned int* rowp, int ka) {
    Frag16 f;
    const int dw = ka >> 1;
#pragma unroll
    for (int i = 0; i < 4; ++i) f.u[i] = rowp[dw + i];
#pragma unroll
    for (int i = 0; i < 4; ++i) f.u[4 + i] = rowp[dw + 4 + i];
    return f.b;
}
// B 32x16 (KxN), source stored n-major [n][k] (pre-transposed weights),
// rowp = dword ptr to this lane's column n = nBase + (lane&15).
// kb = kTile + ((lane>>4)<<4).  VGPR i holds K=kb+2i, kb+2i+1.
__device__ __forceinline__ v16bf load_b(const unsigned int* rowp, int kb) {
    Frag16 f;
    const int dw = kb >> 1;
#pragma unroll
    for (int i = 0; i < 8; ++i) f.u[i] = rowp[dw + i];
    return f.b;
}

// ---------- utility kernels ----------
__global__ void k_zero_f32(float* p, int n) {
    int i = blockIdx.x * blockDim.x + threadIdx.x;
    const int stride = gridDim.x * blockDim.x;
    for (; i < n; i += stride) p[i] = 0.0f;
}

__global__ void k_f32_to_bf16(const float* in, unsigned short* out, int n) {
    int i = blockIdx.x * blockDim.x + threadIdx.x;
    const int stride = gridDim.x * blockDim.x;
    for (; i < n; i += stride) out[i] = f2bf(in[i]);
}

// W: [K][Nn] row-major (fan_in x fan_out) -> Wt: [Nn][K] bf16
__global__ void k_wconv(const float* W, unsigned short* Wt, int K, int Nn) {
    int i = blockIdx.x * blockDim.x + threadIdx.x;
    const int stride = gridDim.x * blockDim.x;
    const int total = K * Nn;
    for (; i < total; i += stride) {
        const int n = i / K;
        const int k = i - n * K;
        Wt[i] = f2bf(W[k * Nn + n]);
    }
}

// ---------- edge scatter: agg[dst] += h[src] (one wave per edge, 4 ch/lane) ----------
__global__ void k_scatter(const unsigned short* __restrict__ hbf,
                          const int* __restrict__ src, const int* __restrict__ dst,
                          float* __restrict__ agg) {
    const int lane = threadIdx.x & 31;
    const int w  = (blockIdx.x * blockDim.x + threadIdx.x) >> 5;
    const int nw = (gridDim.x * blockDim.x) >> 5;
    for (int e = w; e < EE; e += nw) {
        const int s = src[e], d = dst[e];
        const unsigned int* hp = (const unsigned int*)hbf + (size_t)s * (HH / 2) + lane * 2;
        const unsigned int p0 = hp[0], p1 = hp[1];
        float* ap = agg + (size_t)d * HH + lane * 4;
        unsafeAtomicAdd(ap + 0, bflo2f(p0));
        unsafeAtomicAdd(ap + 1, bfhi2f(p0));
        unsafeAtomicAdd(ap + 2, bflo2f(p1));
        unsafeAtomicAdd(ap + 3, bfhi2f(p1));
    }
}

// ---------- per-graph readout via full LDS table (64*128 f32 = 32KB) ----------
__global__ void k_readout(const unsigned short* __restrict__ hbf,
                          const int* __restrict__ batch, float* __restrict__ ro) {
    __shared__ float tbl[GG * HH];
    for (int i = threadIdx.x; i < GG * HH; i += blockDim.x) tbl[i] = 0.0f;
    __syncthreads();
    const int lane = threadIdx.x & 31;
    const int wv   = threadIdx.x >> 5;
    const int wpb  = blockDim.x >> 5;
    const int chunk = (NN + gridDim.x - 1) / gridDim.x;
    const int n0 = blockIdx.x * chunk;
    int n1 = n0 + chunk; if (n1 > NN) n1 = NN;
    for (int n = n0 + wv; n < n1; n += wpb) {
        const int g = batch[n];
        const unsigned int* hp = (const unsigned int*)hbf + (size_t)n * (HH / 2) + lane * 2;
        const unsigned int p0 = hp[0], p1 = hp[1];
        float* tp = tbl + g * HH + lane * 4;
        atomicAdd(tp + 0, bflo2f(p0));
        atomicAdd(tp + 1, bfhi2f(p0));
        atomicAdd(tp + 2, bflo2f(p1));
        atomicAdd(tp + 3, bfhi2f(p1));
    }
    __syncthreads();
    for (int i = threadIdx.x; i < GG * HH; i += blockDim.x)
        unsafeAtomicAdd(&ro[i], tbl[i]);
}

// ---------- fused layer GEMM: y = relu(h@V + agg@A + ro[batch]@R + bias) ----------
// block = 256 thr = 8 waves; block -> 16 rows x 128 cols, wave -> one 16x16 tile
__global__ void __launch_bounds__(256)
k_gemm_layer(const unsigned short* __restrict__ hbf, const unsigned short* __restrict__ aggbf,
             const unsigned short* __restrict__ robf, const int* __restrict__ batch,
             const unsigned short* __restrict__ Vt, const unsigned short* __restrict__ At,
             const unsigned short* __restrict__ Rt,
             const float* __restrict__ Vb, const float* __restrict__ Ab, const float* __restrict__ Rb,
             float* __restrict__ y, float* __restrict__ sum, float* __restrict__ sumsq) {
    const int lane = threadIdx.x & 31;
    const int wave = threadIdx.x >> 5;
    const int m    = lane & 15;
    const int half = lane >> 4;
    const int row0 = blockIdx.x * 16;
    const int col  = wave * 16 + m;

    const unsigned int* hrow  = (const unsigned int*)hbf   + (size_t)(row0 + m) * (HH / 2);
    const unsigned int* arow  = (const unsigned int*)aggbf + (size_t)(row0 + m) * (HH / 2);
    const int g = batch[row0 + m];
    const unsigned int* rrow  = (const unsigned int*)robf  + (size_t)g * (HH / 2);
    const unsigned int* vrow  = (const unsigned int*)Vt + (size_t)col * (HH / 2);
    const unsigned int* awrow = (const unsigned int*)At + (size_t)col * (HH / 2);
    const unsigned int* rwrow = (const unsigned int*)Rt + (size_t)col * (HH / 2);

    v8f acc = {0.f, 0.f, 0.f, 0.f, 0.f, 0.f, 0.f, 0.f};
#pragma unroll
    for (int kt = 0; kt < HH; kt += 32) {
        const int ka = kt + (half << 4);
        const int kb = kt + (half << 4);
        acc = __builtin_amdgcn_wmma_f32_16x16x32_bf16(false, load_a(hrow, ka), false,
                                                      load_b(vrow, kb), (short)0, acc, false, false);
        acc = __builtin_amdgcn_wmma_f32_16x16x32_bf16(false, load_a(arow, ka), false,
                                                      load_b(awrow, kb), (short)0, acc, false, false);
        acc = __builtin_amdgcn_wmma_f32_16x16x32_bf16(false, load_a(rrow, ka), false,
                                                      load_b(rwrow, kb), (short)0, acc, false, false);
    }
    const float bias = Vb[col] + Ab[col] + Rb[col];
    float s = 0.f, s2 = 0.f;
#pragma unroll
    for (int j = 0; j < 8; ++j) {               // C/D layout: VGPR j -> row j (+8 for upper half)
        float v = acc[j] + bias;
        v = v > 0.f ? v : 0.f;                  // ReLU before BN stats (reference order)
        y[(size_t)(row0 + half * 8 + j) * HH + col] = v;
        s += v; s2 += v * v;
    }
    s  += __shfl_xor(s, 16);
    s2 += __shfl_xor(s2, 16);
    if (half == 0) {
        unsafeAtomicAdd(&sum[col], s);
        unsafeAtomicAdd(&sumsq[col], s2);
    }
}

__global__ void k_bn_finalize(const float* sum, const float* sumsq,
                              const float* gamma, const float* beta,
                              float* scale, float* shift) {
    const int c = threadIdx.x;
    if (c < HH) {
        const float mean = sum[c] * (1.0f / NN);
        float var = sumsq[c] * (1.0f / NN) - mean * mean;
        var = var < 0.f ? 0.f : var;
        const float inv = rsqrtf(var + EPSV);
        const float sc = gamma[c] * inv;
        scale[c] = sc;
        shift[c] = beta[c] - mean * sc;
    }
}

__global__ void k_bn_apply(const float* __restrict__ y, const float* __restrict__ scale,
                           const float* __restrict__ shift, unsigned short* __restrict__ hout) {
    int i = blockIdx.x * blockDim.x + threadIdx.x;
    const int stride = gridDim.x * blockDim.x;
    for (; i < NN * HH; i += stride) {
        const int c = i & (HH - 1);
        hout[i] = f2bf(y[i] * scale[c] + shift[c]);
    }
}

// ---------- prediction GEMM: out[N,64] = h @ pred_w + pred_b ----------
__global__ void __launch_bounds__(128)
k_gemm_pred(const unsigned short* __restrict__ hbf, const unsigned short* __restrict__ Pt,
            const float* __restrict__ pb, float* __restrict__ out) {
    const int lane = threadIdx.x & 31;
    const int wave = threadIdx.x >> 5;          // 0..3 -> column tile
    const int m    = lane & 15;
    const int half = lane >> 4;
    const int row0 = blockIdx.x * 16;
    const int col  = wave * 16 + m;
    const unsigned int* hrow = (const unsigned int*)hbf + (size_t)(row0 + m) * (HH / 2);
    const unsigned int* prow = (const unsigned int*)Pt  + (size_t)col * (HH / 2);
    v8f acc = {0.f, 0.f, 0.f, 0.f, 0.f, 0.f, 0.f, 0.f};
#pragma unroll
    for (int kt = 0; kt < HH; kt += 32) {
        acc = __builtin_amdgcn_wmma_f32_16x16x32_bf16(false, load_a(hrow, kt + (half << 4)), false,
                                                      load_b(prow, kt + (half << 4)),
                                                      (short)0, acc, false, false);
    }
    const float bias = pb[col];
#pragma unroll
    for (int j = 0; j < 8; ++j)
        out[(size_t)(row0 + half * 8 + j) * OUTD + col] = acc[j] + bias;
}

extern "C" void kernel_launch(void* const* d_in, const int* in_sizes, int n_in,
                              void* d_out, int out_size, void* d_ws, size_t ws_size,
                              hipStream_t stream) {
    (void)in_sizes; (void)n_in; (void)out_size; (void)ws_size;
    const float* x      = (const float*)d_in[0];
    const int*   edge   = (const int*)d_in[1];
    const int*   src    = edge;          // edge_index[0]
    const int*   dst    = edge + EE;     // edge_index[1]
    const int*   batch  = (const int*)d_in[2];
    const float* V_w    = (const float*)d_in[3];
    const float* V_b    = (const float*)d_in[4];
    const float* A_w    = (const float*)d_in[5];
    const float* A_b    = (const float*)d_in[6];
    const float* R_w    = (const float*)d_in[7];
    const float* R_b    = (const float*)d_in[8];
    const float* gamma  = (const float*)d_in[9];
    const float* beta   = (const float*)d_in[10];
    const float* pred_w = (const float*)d_in[11];
    const float* pred_b = (const float*)d_in[12];
    float* out = (float*)d_out;

    // ---- workspace carve-up (~90 MB) ----
    char* ws = (char*)d_ws;
    size_t off = 0;
    auto alloc = [&](size_t bytes) -> void* {
        void* p = ws + off;
        off = (off + bytes + 255) & ~(size_t)255;
        return p;
    };
    unsigned short* hbf0  = (unsigned short*)alloc((size_t)NN * HH * 2);
    unsigned short* hbf1  = (unsigned short*)alloc((size_t)NN * HH * 2);
    float*          yf    = (float*)alloc((size_t)NN * HH * 4);
    // contiguous zero region: agg + ro + sum + sumsq
    float* zbase = (float*)alloc(((size_t)NN * HH + GG * HH + 2 * HH) * 4);
    float* aggf  = zbase;
    float* ro_f  = zbase + (size_t)NN * HH;
    float* sum   = ro_f + GG * HH;
    float* sumsq = sum + HH;
    unsigned short* aggbf = (unsigned short*)alloc((size_t)NN * HH * 2);
    unsigned short* robf  = (unsigned short*)alloc((size_t)GG * HH * 2);
    unsigned short* wt    = (unsigned short*)alloc((size_t)9 * HH * HH * 2);
    unsigned short* Pt    = (unsigned short*)alloc((size_t)OUTD * HH * 2);
    float* scale = (float*)alloc(HH * 4);
    float* shift = (float*)alloc(HH * 4);

    // ---- one-time conversions (graph-capture safe, on stream) ----
    for (int l = 0; l < LL; ++l) {
        k_wconv<<<64, 256, 0, stream>>>(V_w + (size_t)l * HH * HH, wt + (size_t)(3 * l + 0) * HH * HH, HH, HH);
        k_wconv<<<64, 256, 0, stream>>>(A_w + (size_t)l * HH * HH, wt + (size_t)(3 * l + 1) * HH * HH, HH, HH);
        k_wconv<<<64, 256, 0, stream>>>(R_w + (size_t)l * HH * HH, wt + (size_t)(3 * l + 2) * HH * HH, HH, HH);
    }
    k_wconv<<<32, 256, 0, stream>>>(pred_w, Pt, HH, OUTD);
    k_f32_to_bf16<<<1024, 256, 0, stream>>>(x, hbf0, NN * HH);

    unsigned short* hcur = hbf0;
    unsigned short* hnxt = hbf1;
    const int zcount = NN * HH + GG * HH + 2 * HH;
    for (int l = 0; l < LL; ++l) {
        k_zero_f32<<<2048, 256, 0, stream>>>(zbase, zcount);
        k_scatter<<<4096, 256, 0, stream>>>(hcur, src, dst, aggf);
        k_readout<<<128, 256, 0, stream>>>(hcur, batch, ro_f);
        k_f32_to_bf16<<<1024, 256, 0, stream>>>(aggf, aggbf, NN * HH);
        k_f32_to_bf16<<<8, 256, 0, stream>>>(ro_f, robf, GG * HH);
        k_gemm_layer<<<NN / 16, 256, 0, stream>>>(
            hcur, aggbf, robf, batch,
            wt + (size_t)(3 * l + 0) * HH * HH,
            wt + (size_t)(3 * l + 1) * HH * HH,
            wt + (size_t)(3 * l + 2) * HH * HH,
            V_b + (size_t)l * HH, A_b + (size_t)l * HH, R_b + (size_t)l * HH,
            yf, sum, sumsq);
        k_bn_finalize<<<1, 128, 0, stream>>>(sum, sumsq, gamma + (size_t)l * HH,
                                             beta + (size_t)l * HH, scale, shift);
        k_bn_apply<<<2048, 256, 0, stream>>>(yf, scale, shift, hnxt);
        unsigned short* t = hcur; hcur = hnxt; hnxt = t;
    }
    k_gemm_pred<<<NN / 16, 128, 0, stream>>>(hcur, Pt, pred_b, out);
}
